// DialecticAttention_51805895524527
// MI455X (gfx1250) — compile-verified, matrix operand order checked by reading, not run on hardware
//
#include <hip/hip_runtime.h>
#include <hip/hip_bf16.h>

// ---------------------------------------------------------------------------
// DialecticAttention for MI455X (gfx1250, wave32, WMMA bf16 16x16x32).
// Pipeline: f32->bf16 convert, Q/K/V projection GEMMs (WMMA), RoPE, flash
// attention (WMMA, S^T trick so P-fragment needs no cross-lane transpose),
// output projection GEMM (WMMA), sigmoid confidence head.
// ---------------------------------------------------------------------------

#define DEVI __device__ __forceinline__

constexpr int B_  = 2;
constexpr int L_  = 2048;
constexpr int D_  = 2048;
constexpr int H_  = 16;
constexpr int HD_ = 128;          // head dim
constexpr float SCALE_ = 0.08838834764831845f; // 1/sqrt(128)

typedef __attribute__((ext_vector_type(16))) __bf16          v16bf;
typedef __attribute__((ext_vector_type(8)))  float           v8f;
typedef __attribute__((ext_vector_type(8)))  unsigned short  u16x8;
using us = unsigned short;

DEVI us f2bf(float x) {                      // f32 -> bf16, round-nearest-even
    union { float f; unsigned u; } v; v.f = x;
    unsigned r = v.u + 0x7FFFu + ((v.u >> 16) & 1u);
    return (us)(r >> 16);
}

DEVI v8f zero8() {
    v8f z;
    #pragma unroll
    for (int i = 0; i < 8; ++i) z[i] = 0.0f;
    return z;
}

// One fragment loader for everything (NT layouts): lane t holds row
// (row0 + t&15); halves 0..7 = k + (t>>4)*8, halves 8..15 = k+16 + (t>>4)*8.
// Two contiguous 16B loads per lane.
DEVI v16bf load_frag(const us* base, int ld, int row0, int kb) {
    int t = threadIdx.x & 31;
    const us* p = base + (size_t)(row0 + (t & 15)) * ld + kb + ((t >> 4) << 3);
    union { v16bf v; u16x8 h[2]; } u;
    u.h[0] = *(const u16x8*)p;
    u.h[1] = *(const u16x8*)(p + 16);
    return u.v;
}

DEVI v8f wmma_bf16(v16bf a, v16bf b, v8f c) {
    return __builtin_amdgcn_wmma_f32_16x16x32_bf16(
        false, a, false, b, (short)0, c, false, false);
}

// ---------------------------------------------------------------------------
__global__ void k_f32_to_bf16(const float* __restrict__ s, us* __restrict__ d, int n) {
    int i = blockIdx.x * 256 + threadIdx.x;
    if (i < n) d[i] = f2bf(s[i]);
}

// ---------------------------------------------------------------------------
// C[M,N] (+epilogue) = A[M,K] * B[N,K]^T, bf16 in / f32 accumulate.
// One wave per 32x64 tile. MODE 0: store f32 row-major. MODE 1: store bf16
// transposed into Vt[b,h,d,l].
template <int MODE>
__global__ void __launch_bounds__(32)
k_gemm_nt(const us* __restrict__ A, const us* __restrict__ W,
          float* __restrict__ Cf, us* __restrict__ Cb, int M, int N, int K) {
    int m0 = blockIdx.x * 32;
    int n0 = blockIdx.y * 64;
    v8f acc[2][4];
    #pragma unroll
    for (int i = 0; i < 2; ++i)
        #pragma unroll
        for (int j = 0; j < 4; ++j) acc[i][j] = zero8();

    for (int kb = 0; kb < K; kb += 32) {
        v16bf a0 = load_frag(A, K, m0,      kb);
        v16bf a1 = load_frag(A, K, m0 + 16, kb);
        v16bf b0 = load_frag(W, K, n0,      kb);
        v16bf b1 = load_frag(W, K, n0 + 16, kb);
        v16bf b2 = load_frag(W, K, n0 + 32, kb);
        v16bf b3 = load_frag(W, K, n0 + 48, kb);
        acc[0][0] = wmma_bf16(a0, b0, acc[0][0]);
        acc[0][1] = wmma_bf16(a0, b1, acc[0][1]);
        acc[0][2] = wmma_bf16(a0, b2, acc[0][2]);
        acc[0][3] = wmma_bf16(a0, b3, acc[0][3]);
        acc[1][0] = wmma_bf16(a1, b0, acc[1][0]);
        acc[1][1] = wmma_bf16(a1, b1, acc[1][1]);
        acc[1][2] = wmma_bf16(a1, b2, acc[1][2]);
        acc[1][3] = wmma_bf16(a1, b3, acc[1][3]);
    }

    int t = threadIdx.x & 31;
    int hrow = t >> 4, lo = t & 15;          // C layout: lane=n, reg r -> m
    #pragma unroll
    for (int i = 0; i < 2; ++i)
        #pragma unroll
        for (int j = 0; j < 4; ++j)
            #pragma unroll
            for (int r = 0; r < 8; ++r) {
                int m = m0 + i * 16 + hrow * 8 + r;
                int n = n0 + j * 16 + lo;
                float v = acc[i][j][r];
                if (MODE == 0) {
                    Cf[(size_t)m * N + n] = v;
                } else {                      // Vt[b][h][d][l] bf16
                    int bb = m >> 11, l = m & (L_ - 1);
                    int hh = n >> 7,  d = n & (HD_ - 1);
                    Cb[(((size_t)bb * H_ + hh) * HD_ + d) * L_ + l] = f2bf(v);
                }
            }
}

// ---------------------------------------------------------------------------
// RoPE: read f32 [B,L,D] (D = h*128+d), write bf16 head-major [B,H,L,hd].
__global__ void k_rope(const float* __restrict__ src, us* __restrict__ dst) {
    int i = blockIdx.x * 256 + threadIdx.x;           // over B*L*D, exact grid
    int c = i & (D_ - 1);
    int l = (i >> 11) & (L_ - 1);
    int b = i >> 22;
    int h = c >> 7, d = c & (HD_ - 1);
    int j = d & 63;                                    // freq index
    float f  = (float)l * __expf(-0.14391156831f * (float)j); // ln(1e4)/64
    float cs = cosf(f), sn = sinf(f);
    float q  = src[i];
    float ot = (d < 64) ? -src[i + 64] : src[i - 64];  // rotate_half
    dst[((size_t)(b * H_ + h) * L_ + l) * HD_ + d] = f2bf(q * cs + ot * sn);
}

// ---------------------------------------------------------------------------
// Flash attention, one wave per (b,h,16-q-row tile).
// S^T = K_tile x Q^T so the P-fragment for P@V needs no cross-lane transpose.
__global__ void __launch_bounds__(32)
k_attn(const us* __restrict__ Qh, const us* __restrict__ Kh,
       const us* __restrict__ Vt, us* __restrict__ ctx) {
    int wid = blockIdx.x;
    int qt = wid & (L_ / 16 - 1);
    int h  = (wid >> 7) & (H_ - 1);
    int b  = wid >> 11;
    int t  = threadIdx.x & 31;
    int hrow = t >> 4, lo = t & 15;
    int q0 = qt * 16;

    const us* Qb = Qh + (size_t)(b * H_ + h) * L_ * HD_;
    const us* Kb = Kh + (size_t)(b * H_ + h) * L_ * HD_;
    const us* Vb = Vt + (size_t)(b * H_ + h) * HD_ * L_;

    v16bf qf[4];
    #pragma unroll
    for (int c = 0; c < 4; ++c) qf[c] = load_frag(Qb, HD_, q0, c * 32);

    v8f o[8];
    #pragma unroll
    for (int nt = 0; nt < 8; ++nt) o[nt] = zero8();
    float m_i = -1e30f, l_i = 0.0f;

    for (int j0 = 0; j0 < L_; j0 += 32) {
        v8f st0 = zero8(), st1 = zero8();      // S^T tiles: keys j0.. , j0+16..
        #pragma unroll
        for (int c = 0; c < 4; ++c) {
            v16bf k0 = load_frag(Kb, HD_, j0,      c * 32);
            v16bf k1 = load_frag(Kb, HD_, j0 + 16, c * 32);
            st0 = wmma_bf16(k0, qf[c], st0);
            st1 = wmma_bf16(k1, qf[c], st1);
        }
        // lane t: q = t&15 ; st0 reg r -> k = 8*hrow+r ; st1 -> k = 16+8*hrow+r
        float p0[8], p1[8], mx = -1e30f;
        #pragma unroll
        for (int r = 0; r < 8; ++r) {
            p0[r] = st0[r] * SCALE_; p1[r] = st1[r] * SCALE_;
            mx = fmaxf(mx, fmaxf(p0[r], p1[r]));
        }
        mx = fmaxf(mx, __shfl_xor(mx, 16, 32));       // row max over 32 keys
        float m_new = fmaxf(m_i, mx);
        float alpha = __expf(m_i - m_new);
        float rsum = 0.0f;
        #pragma unroll
        for (int r = 0; r < 8; ++r) {
            p0[r] = __expf(p0[r] - m_new);
            p1[r] = __expf(p1[r] - m_new);
            rsum += p0[r] + p1[r];
        }
        rsum += __shfl_xor(rsum, 16, 32);
        l_i = l_i * alpha + rsum;
        m_i = m_new;
        #pragma unroll
        for (int r = 0; r < 8; ++r) {                 // rescale O rows (q=reg)
            float a = __shfl(alpha, hrow * 8 + r, 32);
            #pragma unroll
            for (int nt = 0; nt < 8; ++nt) o[nt][r] *= a;
        }
        // Pack P A-fragment: halves 0..7 <- st0 regs, 8..15 <- st1 regs.
        union { v16bf v; us e[16]; } pf;
        #pragma unroll
        for (int r = 0; r < 8; ++r) { pf.e[r] = f2bf(p0[r]); pf.e[8 + r] = f2bf(p1[r]); }
        #pragma unroll
        for (int nt = 0; nt < 8; ++nt) {
            v16bf vf = load_frag(Vb, L_, nt * 16, j0);   // Vt rows contiguous in k
            o[nt] = wmma_bf16(pf.v, vf, o[nt]);
        }
    }

    float linv = 1.0f / l_i;
    #pragma unroll
    for (int r = 0; r < 8; ++r) {
        float s = __shfl(linv, hrow * 8 + r, 32);
        int row = q0 + hrow * 8 + r;
        size_t base = ((size_t)(b * L_ + row)) * D_ + h * HD_;
        #pragma unroll
        for (int nt = 0; nt < 8; ++nt)
            ctx[base + nt * 16 + lo] = f2bf(o[nt][r] * s);
    }
}

// ---------------------------------------------------------------------------
__global__ void k_conf(const float* __restrict__ out, const float* __restrict__ Wc,
                       float* __restrict__ conf, float* __restrict__ logits) {
    int row  = blockIdx.x * 8 + (threadIdx.x >> 5);
    int lane = threadIdx.x & 31;
    float s = 0.0f;
    const float* p = out + (size_t)row * D_;
    for (int i = lane; i < D_; i += 32) s += p[i] * Wc[i];
    #pragma unroll
    for (int off = 16; off; off >>= 1) s += __shfl_xor(s, off, 32);
    if (lane == 0) {
        logits[row] = s;
        conf[row]   = 1.0f / (1.0f + __expf(-s));
    }
}

// ---------------------------------------------------------------------------
extern "C" void kernel_launch(void* const* d_in, const int* in_sizes, int n_in,
                              void* d_out, int out_size, void* d_ws, size_t ws_size,
                              hipStream_t stream) {
    const float* x  = (const float*)d_in[0];
    const float* Wq = (const float*)d_in[1];
    const float* Wk = (const float*)d_in[2];
    const float* Wv = (const float*)d_in[3];
    const float* Wo = (const float*)d_in[4];
    const float* Wc = (const float*)d_in[5];

    // Workspace layout (stream-ordered aliasing; 112 MB total):
    //  [  0MB,16MB) xb   -> later Qh   (xb dead after V GEMM)
    //  [ 16MB,32MB) Wqb,Wkb -> later Vt (dead after K GEMM)
    //  [ 32MB,48MB) Wvb,Wob
    //  [ 48MB,80MB) Qf   -> later Kh (48..64) + ctx (64..80)
    //  [ 80MB,112MB) Kf
    unsigned char* w = (unsigned char*)d_ws;
    us*    xb   = (us*)(w);
    us*    Qh   = (us*)(w);
    us*    Wqb  = (us*)(w + (16u << 20));
    us*    Wkb  = (us*)(w + (24u << 20));
    us*    Vt   = (us*)(w + (16u << 20));
    us*    Wvb  = (us*)(w + (32u << 20));
    us*    Wob  = (us*)(w + (40u << 20));
    float* Qf   = (float*)(w + (48u << 20));
    us*    Kh   = (us*)(w + (48u << 20));
    us*    ctx  = (us*)(w + (64u << 20));
    float* Kf   = (float*)(w + (80u << 20));

    float* out    = (float*)d_out;
    float* conf   = out + (size_t)B_ * L_ * D_;
    float* logits = conf + (size_t)B_ * L_;

    const int M = B_ * L_;                  // 4096
    const int nx = B_ * L_ * D_;            // 8388608
    const int nw = D_ * D_;                 // 4194304
    dim3 ggrid(M / 32, D_ / 64);            // 128 x 32 waves

    // 1) convert to bf16
    k_f32_to_bf16<<<(nx + 255) / 256, 256, 0, stream>>>(x,  xb,  nx);
    k_f32_to_bf16<<<(nw + 255) / 256, 256, 0, stream>>>(Wq, Wqb, nw);
    k_f32_to_bf16<<<(nw + 255) / 256, 256, 0, stream>>>(Wk, Wkb, nw);
    k_f32_to_bf16<<<(nw + 255) / 256, 256, 0, stream>>>(Wv, Wvb, nw);
    k_f32_to_bf16<<<(nw + 255) / 256, 256, 0, stream>>>(Wo, Wob, nw);

    // 2) projections (WMMA)
    k_gemm_nt<0><<<ggrid, 32, 0, stream>>>(xb, Wqb, Qf, nullptr, M, D_, D_);
    k_gemm_nt<0><<<ggrid, 32, 0, stream>>>(xb, Wkb, Kf, nullptr, M, D_, D_);
    k_gemm_nt<1><<<ggrid, 32, 0, stream>>>(xb, Wvb, nullptr, Vt, M, D_, D_);

    // 3) RoPE -> head-major bf16
    k_rope<<<nx / 256, 256, 0, stream>>>(Qf, Qh);
    k_rope<<<nx / 256, 256, 0, stream>>>(Kf, Kh);

    // 4) flash attention (WMMA)
    k_attn<<<B_ * H_ * (L_ / 16), 32, 0, stream>>>(Qh, Kh, Vt, ctx);

    // 5) output projection (WMMA) straight into d_out
    k_gemm_nt<0><<<ggrid, 32, 0, stream>>>(ctx, Wob, out, nullptr, M, D_, D_);

    // 6) confidence head
    k_conf<<<(B_ * L_) / 8, 256, 0, stream>>>(out, Wc, conf, logits);
}
